// EGALayer_17042430230828
// MI455X (gfx1250) — matrix operand order repaired and checked by reading, not compiled, not run on hardware
//
#include <hip/hip_runtime.h>
#include <math.h>

#define N_NODES 50000
#define N_EDGES 800000
#define IN_DIM  256
#define OUT_DIM 64
#define HEADS   8
#define NCAT    (HEADS * OUT_DIM)   // 512 concatenated output columns

typedef __bf16 bf16;
typedef __attribute__((ext_vector_type(8)))  __bf16 v8bf;
typedef __attribute__((ext_vector_type(16))) __bf16 v16bf;
typedef __attribute__((ext_vector_type(8)))  float  v8f;

// ---------------------------------------------------------------------------
// helpers
// ---------------------------------------------------------------------------
__device__ __forceinline__ unsigned fkey(float f) {   // order-preserving uint
  unsigned u = __float_as_uint(f);
  return (u & 0x80000000u) ? ~u : (u | 0x80000000u);
}
__device__ __forceinline__ float fkey_inv(unsigned k) {
  return (k & 0x80000000u) ? __uint_as_float(k & 0x7fffffffu)
                           : __uint_as_float(~k);
}

// A fragment (16x32 bf16, ISA 7.12.2): lanes 0-15 = rows, K chunks {ksel..ksel+7}
// and {16+ksel..16+ksel+7} with ksel = 8*(lane>=16).
__device__ __forceinline__ v16bf load_fragA(const bf16* rowp, int kk, int ksel) {
  v8bf c0 = *(const v8bf*)(rowp + kk + ksel);
  v8bf c1 = *(const v8bf*)(rowp + kk + 16 + ksel);
  return __builtin_shufflevector(c0, c1, 0,1,2,3,4,5,6,7,8,9,10,11,12,13,14,15);
}
// B fragment (32x16 bf16): lane = column, K chunks {khalf..khalf+7} and
// {16+khalf..16+khalf+7} with khalf = 8*(lane>=16), from [N][K] transposed W.
__device__ __forceinline__ v16bf load_fragB(const bf16* colp, int kk, int khalf) {
  v8bf c0 = *(const v8bf*)(colp + kk + khalf);
  v8bf c1 = *(const v8bf*)(colp + kk + 16 + khalf);
  return __builtin_shufflevector(c0, c1, 0,1,2,3,4,5,6,7,8,9,10,11,12,13,14,15);
}

// ---------------------------------------------------------------------------
// prep kernels
// ---------------------------------------------------------------------------
__global__ __launch_bounds__(256) void k_zero_init(float* out, unsigned* gmax, float* sums) {
  size_t i = (size_t)blockIdx.x * blockDim.x + threadIdx.x;
  if (i < (size_t)N_NODES * OUT_DIM) out[i] = 0.0f;
  if (i < HEADS) { gmax[i] = 0u; sums[i] = 0.0f; }
}

__global__ __launch_bounds__(256) void k_split_x(const float* x, bf16* xhi, bf16* xlo) {
  size_t i = (size_t)blockIdx.x * blockDim.x + threadIdx.x;
  if (i >= (size_t)N_NODES * IN_DIM) return;
  float v  = x[i];
  bf16  hi = (bf16)v;
  xhi[i] = hi;
  xlo[i] = (bf16)(v - (float)hi);
}

// W[h][d][o] -> Wt[n = h*64+o][d]  (bf16 hi/lo), so B loads are K-contiguous
__global__ __launch_bounds__(256) void k_split_w(const float* W, bf16* wthi, bf16* wtlo) {
  int i = blockIdx.x * blockDim.x + threadIdx.x;
  if (i >= HEADS * IN_DIM * OUT_DIM) return;
  int h   = i / (IN_DIM * OUT_DIM);
  int rem = i % (IN_DIM * OUT_DIM);
  int d   = rem / OUT_DIM;
  int o   = rem % OUT_DIM;
  int n   = h * OUT_DIM + o;
  float v  = W[i];
  bf16  hi = (bf16)v;
  wthi[(size_t)n * IN_DIM + d] = hi;
  wtlo[(size_t)n * IN_DIM + d] = (bf16)(v - (float)hi);
}

// ---------------------------------------------------------------------------
// GEMM: Wh[n][col] = x[n][:] . Wt[col][:] + b[col]   (bf16 split-3 WMMA)
// block = 8 waves; wave computes a 16(M) x 64(N) strip (4 accumulator tiles)
// ---------------------------------------------------------------------------
__global__ __launch_bounds__(256) void k_gemm(const bf16* __restrict__ xhi,
                                              const bf16* __restrict__ xlo,
                                              const bf16* __restrict__ wthi,
                                              const bf16* __restrict__ wtlo,
                                              const float* __restrict__ bvec,
                                              float* __restrict__ Wh) {
  const int lane  = threadIdx.x & 31;
  const int wave  = threadIdx.x >> 5;
  const int m0    = blockIdx.y * 128 + wave * 16;
  const int n0    = blockIdx.x * 64;
  const int r     = lane & 15;
  const int ksel  = (lane >> 4) * 8;    // A half-select
  const int khalf = (lane >> 4) * 8;    // B half-select

  int row = m0 + r;
  if (row >= N_NODES) row = N_NODES - 1;           // clamp; stores are guarded
  const bf16* ahp = xhi + (size_t)row * IN_DIM;
  const bf16* alp = xlo + (size_t)row * IN_DIM;

  const bf16* bh[4];
  const bf16* bl[4];
#pragma unroll
  for (int t = 0; t < 4; ++t) {
    int col = n0 + t * 16 + r;
    bh[t] = wthi + (size_t)col * IN_DIM;
    bl[t] = wtlo + (size_t)col * IN_DIM;
  }

  v8f acc[4] = {};
#pragma unroll
  for (int kk = 0; kk < IN_DIM; kk += 32) {
    v16bf a_hi = load_fragA(ahp, kk, ksel);
    v16bf a_lo = load_fragA(alp, kk, ksel);
#pragma unroll
    for (int t = 0; t < 4; ++t) {
      v16bf b_hi = load_fragB(bh[t], kk, khalf);
      v16bf b_lo = load_fragB(bl[t], kk, khalf);
      acc[t] = __builtin_amdgcn_wmma_f32_16x16x32_bf16(false, a_hi, false, b_hi,
                                                       (short)0, acc[t], false, false);
      acc[t] = __builtin_amdgcn_wmma_f32_16x16x32_bf16(false, a_lo, false, b_hi,
                                                       (short)0, acc[t], false, false);
      acc[t] = __builtin_amdgcn_wmma_f32_16x16x32_bf16(false, a_hi, false, b_lo,
                                                       (short)0, acc[t], false, false);
    }
  }

  // C/D layout: VGPR rr -> M = rr (lanes 0-15) or rr+8 (lanes 16-31); N = lane&15
  const int rbase = m0 + ((lane >> 4) << 3);
#pragma unroll
  for (int t = 0; t < 4; ++t) {
    int   col  = n0 + t * 16 + r;
    float bias = bvec[col];
#pragma unroll
    for (int rr = 0; rr < 8; ++rr) {
      int mrow = rbase + rr;
      if (mrow < N_NODES)
        Wh[(size_t)mrow * NCAT + col] = acc[t][rr] + bias;
    }
  }
}

// ---------------------------------------------------------------------------
// per-node attention dots: s_src[n][h] = <Wh[n,h,:], a[h,:64]>, s_dst with a[h,64:]
// one wave per (node, head)
// ---------------------------------------------------------------------------
__global__ __launch_bounds__(256) void k_scores(const float* __restrict__ Wh,
                                                const float* __restrict__ a,
                                                float* __restrict__ ssrc,
                                                float* __restrict__ sdst) {
  int n    = blockIdx.x;
  int h    = threadIdx.x >> 5;
  int lane = threadIdx.x & 31;
  const float* wrow = Wh + (size_t)n * NCAT + h * OUT_DIM;
  const float* ah   = a + h * (2 * OUT_DIM);
  float v0 = wrow[lane], v1 = wrow[lane + 32];
  float ps = v0 * ah[lane]      + v1 * ah[lane + 32];
  float pd = v0 * ah[64 + lane] + v1 * ah[96 + lane];
#pragma unroll
  for (int off = 16; off > 0; off >>= 1) {
    ps += __shfl_xor(ps, off);
    pd += __shfl_xor(pd, off);
  }
  if (lane == 0) {
    ssrc[(size_t)n * HEADS + h] = ps;
    sdst[(size_t)n * HEADS + h] = pd;
  }
}

// ---------------------------------------------------------------------------
// edge pass 1: e[h] = leaky_relu(s_src[row,h] + s_dst[col,h]); global per-head max
// ---------------------------------------------------------------------------
__global__ __launch_bounds__(256) void k_edge1(const int* __restrict__ ei,
                                               const float* __restrict__ ssrc,
                                               const float* __restrict__ sdst,
                                               float* __restrict__ ebuf,
                                               unsigned* __restrict__ gmax) {
  __shared__ unsigned smax[HEADS];
  if (threadIdx.x < HEADS) smax[threadIdx.x] = 0u;
  __syncthreads();

  int e = blockIdx.x * blockDim.x + threadIdx.x;
  unsigned kv[HEADS];
  if (e < N_EDGES) {
    int r = ei[e];
    int c = ei[N_EDGES + e];
    const float* sr = ssrc + (size_t)r * HEADS;
    const float* sd = sdst + (size_t)c * HEADS;
    float ev[HEADS];
#pragma unroll
    for (int h = 0; h < HEADS; ++h) {
      float t = sr[h] + sd[h];
      t = t > 0.0f ? t : 0.01f * t;   // leaky_relu(0.01)
      ev[h] = t;
      kv[h] = fkey(t);
    }
    float4* op = (float4*)(ebuf + (size_t)e * HEADS);
    op[0] = make_float4(ev[0], ev[1], ev[2], ev[3]);
    op[1] = make_float4(ev[4], ev[5], ev[6], ev[7]);
  } else {
#pragma unroll
    for (int h = 0; h < HEADS; ++h) kv[h] = 0u;
  }

#pragma unroll
  for (int h = 0; h < HEADS; ++h)
#pragma unroll
    for (int off = 16; off > 0; off >>= 1) {
      unsigned o = __shfl_xor(kv[h], off);
      kv[h] = (kv[h] > o) ? kv[h] : o;
    }
  if ((threadIdx.x & 31) == 0)
    for (int h = 0; h < HEADS; ++h) atomicMax(&smax[h], kv[h]);
  __syncthreads();
  if (threadIdx.x < HEADS) atomicMax(&gmax[threadIdx.x], smax[threadIdx.x]);
}

// ---------------------------------------------------------------------------
// edge pass 2: p = exp(e - max[h]) (in place); global per-head sum
// ---------------------------------------------------------------------------
__global__ __launch_bounds__(256) void k_edge2(float* __restrict__ ebuf,
                                               const unsigned* __restrict__ gmax,
                                               float* __restrict__ sums) {
  __shared__ float ssum[HEADS];
  if (threadIdx.x < HEADS) ssum[threadIdx.x] = 0.0f;
  __syncthreads();

  float m[HEADS];
#pragma unroll
  for (int h = 0; h < HEADS; ++h) m[h] = fkey_inv(gmax[h]);

  int e = blockIdx.x * blockDim.x + threadIdx.x;
  float pv[HEADS];
  if (e < N_EDGES) {
    float4* ep = (float4*)(ebuf + (size_t)e * HEADS);
    float4 e0 = ep[0], e1 = ep[1];
    pv[0] = expf(e0.x - m[0]); pv[1] = expf(e0.y - m[1]);
    pv[2] = expf(e0.z - m[2]); pv[3] = expf(e0.w - m[3]);
    pv[4] = expf(e1.x - m[4]); pv[5] = expf(e1.y - m[5]);
    pv[6] = expf(e1.z - m[6]); pv[7] = expf(e1.w - m[7]);
    ep[0] = make_float4(pv[0], pv[1], pv[2], pv[3]);
    ep[1] = make_float4(pv[4], pv[5], pv[6], pv[7]);
  } else {
#pragma unroll
    for (int h = 0; h < HEADS; ++h) pv[h] = 0.0f;
  }

#pragma unroll
  for (int h = 0; h < HEADS; ++h)
#pragma unroll
    for (int off = 16; off > 0; off >>= 1) pv[h] += __shfl_xor(pv[h], off);
  if ((threadIdx.x & 31) == 0)
    for (int h = 0; h < HEADS; ++h) atomicAdd(&ssum[h], pv[h]);
  __syncthreads();
  if (threadIdx.x < HEADS) atomicAdd(&sums[threadIdx.x], ssum[threadIdx.x]);
}

// ---------------------------------------------------------------------------
// coef[h] = softmax(gate)[h] / sum[h]   (single wave)
// ---------------------------------------------------------------------------
__global__ void k_coef(const float* gate, const float* sums, float* coef) {
  int t = threadIdx.x;
  float g = (t < HEADS) ? gate[t] : -1.0e30f;
  float gm = g;
#pragma unroll
  for (int off = 16; off > 0; off >>= 1) gm = fmaxf(gm, __shfl_xor(gm, off));
  float eg = (t < HEADS) ? expf(g - gm) : 0.0f;
  float tot = eg;
#pragma unroll
  for (int off = 16; off > 0; off >>= 1) tot += __shfl_xor(tot, off);
  if (t < HEADS) coef[t] = (eg / tot) / sums[t];
}

// ---------------------------------------------------------------------------
// edge pass 3: out[row,o] += sum_h coef[h]*p[h,e]*Wh[col, h*64+o]
// one wave per edge (grid-stride); heads combined before the atomic
// ---------------------------------------------------------------------------
__global__ __launch_bounds__(256) void k_edge3(const int* __restrict__ ei,
                                               const float* __restrict__ ebuf,
                                               const float* __restrict__ coef,
                                               const float* __restrict__ Wh,
                                               float* __restrict__ out) {
  int lane   = threadIdx.x & 31;
  int wid    = (int)((blockIdx.x * (unsigned)blockDim.x + threadIdx.x) >> 5);
  int nwaves = (int)((gridDim.x * blockDim.x) >> 5);

  float cf[HEADS];
#pragma unroll
  for (int h = 0; h < HEADS; ++h) cf[h] = coef[h];

  for (int e = wid; e < N_EDGES; e += nwaves) {
    int r = ei[e];
    int c = ei[N_EDGES + e];
    const float4* pp = (const float4*)(ebuf + (size_t)e * HEADS);
    float4 p0 = pp[0], p1 = pp[1];
    float w[HEADS] = {p0.x * cf[0], p0.y * cf[1], p0.z * cf[2], p0.w * cf[3],
                      p1.x * cf[4], p1.y * cf[5], p1.z * cf[6], p1.w * cf[7]};
    const float* wrow = Wh + (size_t)c * NCAT;
    float a0 = 0.0f, a1 = 0.0f;
#pragma unroll
    for (int h = 0; h < HEADS; ++h) {
      a0 = fmaf(w[h], wrow[h * OUT_DIM + lane],      a0);
      a1 = fmaf(w[h], wrow[h * OUT_DIM + 32 + lane], a1);
    }
    float* orow = out + (size_t)r * OUT_DIM;
    __hip_atomic_fetch_add(&orow[lane],      a0, __ATOMIC_RELAXED, __HIP_MEMORY_SCOPE_AGENT);
    __hip_atomic_fetch_add(&orow[lane + 32], a1, __ATOMIC_RELAXED, __HIP_MEMORY_SCOPE_AGENT);
  }
}

// ---------------------------------------------------------------------------
extern "C" void kernel_launch(void* const* d_in, const int* in_sizes, int n_in,
                              void* d_out, int out_size, void* d_ws, size_t ws_size,
                              hipStream_t stream) {
  (void)in_sizes; (void)n_in; (void)out_size; (void)ws_size;
  const float* x    = (const float*)d_in[0];
  const int*   ei   = (const int*)d_in[1];
  const float* W    = (const float*)d_in[2];
  const float* b    = (const float*)d_in[3];
  const float* a    = (const float*)d_in[4];
  const float* gate = (const float*)d_in[5];
  float* out = (float*)d_out;

  char* p = (char*)d_ws;
  float* Wh   = (float*)p;  p += (size_t)N_NODES * NCAT * sizeof(float);    // 102.4 MB
  bf16*  xhi  = (bf16*)p;   p += (size_t)N_NODES * IN_DIM * sizeof(bf16);   // 25.6 MB
  bf16*  xlo  = (bf16*)p;   p += (size_t)N_NODES * IN_DIM * sizeof(bf16);   // 25.6 MB
  bf16*  wthi = (bf16*)p;   p += (size_t)NCAT * IN_DIM * sizeof(bf16);      // 0.26 MB
  bf16*  wtlo = (bf16*)p;   p += (size_t)NCAT * IN_DIM * sizeof(bf16);      // 0.26 MB
  float* ssrc = (float*)p;  p += (size_t)N_NODES * HEADS * sizeof(float);   // 1.6 MB
  float* sdst = (float*)p;  p += (size_t)N_NODES * HEADS * sizeof(float);   // 1.6 MB
  float* ebuf = (float*)p;  p += (size_t)N_EDGES * HEADS * sizeof(float);   // 25.6 MB
  unsigned* gmax = (unsigned*)p; p += 64;
  float* sums = (float*)p;  p += 64;
  float* coef = (float*)p;  p += 64;

  k_zero_init<<<(N_NODES * OUT_DIM + 255) / 256, 256, 0, stream>>>(out, gmax, sums);
  k_split_x<<<(N_NODES * IN_DIM + 255) / 256, 256, 0, stream>>>(x, xhi, xlo);
  k_split_w<<<(HEADS * IN_DIM * OUT_DIM + 255) / 256, 256, 0, stream>>>(W, wthi, wtlo);

  dim3 ggrid(NCAT / 64, (N_NODES + 127) / 128);
  k_gemm<<<ggrid, 256, 0, stream>>>(xhi, xlo, wthi, wtlo, b, Wh);

  k_scores<<<N_NODES, 256, 0, stream>>>(Wh, a, ssrc, sdst);
  k_edge1<<<(N_EDGES + 255) / 256, 256, 0, stream>>>(ei, ssrc, sdst, ebuf, gmax);
  k_edge2<<<(N_EDGES + 255) / 256, 256, 0, stream>>>(ebuf, gmax, sums);
  k_coef<<<1, 32, 0, stream>>>(gate, sums, coef);
  k_edge3<<<6250, 256, 0, stream>>>(ei, ebuf, coef, Wh, out);
}
